// SingleHeadCausalAttention_23364622090633
// MI455X (gfx1250) — matrix operand order repaired and checked by reading, not compiled, not run on hardware
//
#include <hip/hip_runtime.h>
#include <hip/hip_bf16.h>
#include <math.h>

// ---------------------------------------------------------------------------
// Single-head causal attention for MI455X (gfx1250, wave32, WMMA bf16).
//   B=8, T=2048, C=1024, D=64.
// Pipeline:
//   1) repack Wq/Wk/Wv (f32) -> bf16 in WMMA B-operand register layout
//   2) projection: one wave per (16-row tile, matrix) pair; A (x) and B
//      (weights) register tiles double-buffered across k-steps so loads hide
//      behind v_wmma_f32_16x16x32_bf16 bursts. Q pre-scaled by 1/sqrt(D);
//      K written row-major bf16; V written transposed bf16.
//   3) flash attention: per-wave 16-query tile, online softmax (f32 state);
//      K operands double-buffered across key tiles, V operands hoisted ahead
//      of softmax, tile kt+2 prefetched (global_prefetch_b8); P relaid
//      C/D-layout -> A-layout through a per-wave LDS tile.
// ---------------------------------------------------------------------------

typedef __attribute__((ext_vector_type(16))) __bf16 v16bf;
typedef __attribute__((ext_vector_type(8)))  float  v8f;

#define B_BATCH 8
#define T_SEQ   2048
#define C_DIM   1024
#define D_HEAD  64
#define KSTEPS  (C_DIM / 32)   // 32 k-steps for projection
#define DTILES  (D_HEAD / 16)  // 4 column tiles of the head dim

// ---------------------------------------------------------------------------
// Kernel 0: repack W (C x 64, f32 row-major) into bf16 B-operand layout.
// B-matrix 32x16 bf16 k order used here (wave32): lane L -> column n = L&15,
// halves j=0..15 -> K = (L>>4)*16 + j. Packed so the projection kernel loads
// one contiguous 32-byte v16bf per lane:
//   wpack[((mat*32 + ks)*4 + dt)*512 + lane*16 + j]
// ---------------------------------------------------------------------------
__global__ void repack_w_kernel(const float* __restrict__ Wq,
                                const float* __restrict__ Wk,
                                const float* __restrict__ Wv,
                                __bf16* __restrict__ wpack) {
  const int tid   = blockIdx.x * blockDim.x + threadIdx.x;
  const int total = 3 * KSTEPS * DTILES * 512;
  if (tid >= total) return;
  const int j    =  tid        & 15;
  const int lane = (tid >> 4)  & 31;
  const int dt   = (tid >> 9)  & 3;
  const int ks   = (tid >> 11) & 31;
  const int mat  =  tid >> 16;
  const float* W = (mat == 0) ? Wq : (mat == 1) ? Wk : Wv;
  const int n    = lane & 15;
  const int kb2  = (lane >> 4) * 16;
  const int krow = ks * 32 + kb2 + j;
  wpack[tid] = (__bf16)W[krow * D_HEAD + dt * 16 + n];
}

// ---------------------------------------------------------------------------
// Kernel 1: projection, one wave per (row-tile, matrix): 3072 waves.
// A-matrix 16x32 bf16 k order used here (same as B): lane L -> row = L&15,
//   halves j=0..15 -> K = (L>>4)*16 + j
// C/D layout: lane L, VGPR r -> element [row0 + r + 8*(L>>4)][dt*16 + (L&15)]
// A and B register tiles are software double-buffered across k-steps.
// ---------------------------------------------------------------------------
__global__ __launch_bounds__(128, 1) void proj_kernel(
    const float*  __restrict__ x,
    const __bf16* __restrict__ wpack,
    const float*  __restrict__ bq,
    const float*  __restrict__ bk,
    const float*  __restrict__ bv,
    __bf16* __restrict__ q_bf,    // [B*T][64] row-major, pre-scaled by 0.125
    __bf16* __restrict__ k_bf,    // [B*T][64] row-major
    __bf16* __restrict__ vt_bf) { // [B][64][T] transposed
  const int lane = threadIdx.x & 31;
  const int wid  = blockIdx.x * 4 + (threadIdx.x >> 5);  // 0..3071
  const int mat  = wid % 3;                               // 0=Q 1=K 2=V
  const int tile = wid / 3;                               // 0..1023
  const int n    = lane & 15;
  const int hi   = lane >> 4;
  const int row0 = tile * 16;            // global row in [B*T]
  const int rowA = row0 + n;
  const int kbA  = hi * 16;

  const float* bias = (mat == 0) ? bq : (mat == 1) ? bk : bv;
  v8f acc[DTILES];
#pragma unroll
  for (int dt = 0; dt < DTILES; ++dt) {
    const float bb = bias[dt * 16 + n];
#pragma unroll
    for (int r = 0; r < 8; ++r) acc[dt][r] = bb;
  }

  const __bf16* wb = wpack + (((size_t)mat * KSTEPS * DTILES) << 9) + lane * 16;
  const float*  xr = x + (size_t)rowA * C_DIM;

  v16bf Ba[DTILES], Bb[DTILES];
  float4 Aa[4], Ab[4];

  // preload stage 0
#pragma unroll
  for (int dt = 0; dt < DTILES; ++dt)
    Ba[dt] = *(const v16bf*)(wb + ((size_t)(0 * DTILES + dt) << 9));
  Aa[0] = *(const float4*)(xr + kbA);
  Aa[1] = *(const float4*)(xr + kbA + 4);
  Aa[2] = *(const float4*)(xr + kbA + 8);
  Aa[3] = *(const float4*)(xr + kbA + 12);

  auto step = [&](int ks, v16bf (&Bcur)[DTILES], float4 (&Acur)[4],
                  v16bf (&Bnxt)[DTILES], float4 (&Anxt)[4]) {
    // issue next stage's loads first (a full iteration ahead of their use)
    if (ks + 1 < KSTEPS) {
#pragma unroll
      for (int dt = 0; dt < DTILES; ++dt)
        Bnxt[dt] = *(const v16bf*)(wb + ((size_t)((ks + 1) * DTILES + dt) << 9));
      const float* xn = xr + (ks + 1) * 32;
      Anxt[0] = *(const float4*)(xn + kbA);
      Anxt[1] = *(const float4*)(xn + kbA + 4);
      Anxt[2] = *(const float4*)(xn + kbA + 8);
      Anxt[3] = *(const float4*)(xn + kbA + 12);
    }
    // convert current A tile to bf16
    v16bf A;
    A[0]  = (__bf16)Acur[0].x; A[1]  = (__bf16)Acur[0].y;
    A[2]  = (__bf16)Acur[0].z; A[3]  = (__bf16)Acur[0].w;
    A[4]  = (__bf16)Acur[1].x; A[5]  = (__bf16)Acur[1].y;
    A[6]  = (__bf16)Acur[1].z; A[7]  = (__bf16)Acur[1].w;
    A[8]  = (__bf16)Acur[2].x; A[9]  = (__bf16)Acur[2].y;
    A[10] = (__bf16)Acur[2].z; A[11] = (__bf16)Acur[2].w;
    A[12] = (__bf16)Acur[3].x; A[13] = (__bf16)Acur[3].y;
    A[14] = (__bf16)Acur[3].z; A[15] = (__bf16)Acur[3].w;
    // 4 WMMAs on already-resident operands
#pragma unroll
    for (int dt = 0; dt < DTILES; ++dt)
      acc[dt] = __builtin_amdgcn_wmma_f32_16x16x32_bf16(false, A, false, Bcur[dt],
                                                        (short)0, acc[dt], false, false);
  };

  for (int ks = 0;;) {
    step(ks, Ba, Aa, Bb, Ab);
    if (++ks >= KSTEPS) break;
    step(ks, Bb, Ab, Ba, Aa);
    if (++ks >= KSTEPS) break;
  }

  // ---- epilogue: store this wave's matrix --------------------------------
  const int b       = row0 >> 11;      // 2048 rows per batch
  const int klocal0 = row0 & 2047;
  if (mat == 0) {
#pragma unroll
    for (int dt = 0; dt < DTILES; ++dt)
#pragma unroll
      for (int r = 0; r < 8; ++r)
        q_bf[(size_t)(row0 + r + 8 * hi) * D_HEAD + dt * 16 + n] =
            (__bf16)(acc[dt][r] * 0.125f);           // fold 1/sqrt(64)
  } else if (mat == 1) {
#pragma unroll
    for (int dt = 0; dt < DTILES; ++dt)
#pragma unroll
      for (int r = 0; r < 8; ++r)
        k_bf[(size_t)(row0 + r + 8 * hi) * D_HEAD + dt * 16 + n] = (__bf16)acc[dt][r];
  } else {
#pragma unroll
    for (int dt = 0; dt < DTILES; ++dt)
#pragma unroll
      for (int r = 0; r < 8; ++r)
        vt_bf[((size_t)b * D_HEAD + dt * 16 + n) * T_SEQ + klocal0 + r + 8 * hi] =
            (__bf16)acc[dt][r];
  }
}

// ---------------------------------------------------------------------------
// Kernel 2: flash attention. One wave per 16-query tile (1024 tiles total).
// Per 32-key tile: 4 QK^T WMMAs + 4 P@V WMMAs; K operands double-buffered
// across tiles; V operands loaded at iteration top (used after softmax);
// P goes through per-wave LDS tile for the C/D -> A relayout.
// ---------------------------------------------------------------------------
__global__ __launch_bounds__(128, 1) void attn_kernel(
    const __bf16* __restrict__ q_bf,
    const __bf16* __restrict__ k_bf,
    const __bf16* __restrict__ vt_bf,
    float* __restrict__ out) {
  __shared__ __bf16 Pl[4][16][32];     // per-wave P tile (16 rows x 32 keys)
  const int lane = threadIdx.x & 31;
  const int w    = threadIdx.x >> 5;
  const int wid  = blockIdx.x * 4 + w; // 0..1023
  const int b    = wid >> 7;           // 128 q-tiles per batch
  const int qt   = wid & 127;
  const int q0   = qt * 16;            // within batch
  const int n    = lane & 15;
  const int hi   = lane >> 4;
  const size_t g0 = (size_t)b * T_SEQ + q0;

  // Q tile in A-layout (two K-chunks of 32 head dims)
  v16bf qA[2];
#pragma unroll
  for (int kc = 0; kc < 2; ++kc) {
    const __bf16* qp = q_bf + (g0 + n) * D_HEAD + kc * 32;
#pragma unroll
    for (int j = 0; j < 8; ++j) {
      qA[kc][j]     = qp[hi * 16 + j];
      qA[kc][8 + j] = qp[hi * 16 + 8 + j];
    }
  }

  v8f acc[DTILES];
  float m[8], l[8];
#pragma unroll
  for (int dt = 0; dt < DTILES; ++dt)
#pragma unroll
    for (int r = 0; r < 8; ++r) acc[dt][r] = 0.f;
#pragma unroll
  for (int r = 0; r < 8; ++r) { m[r] = -3.0e38f; l[r] = 0.f; }

  const int nkt = (q0 + 16 + 31) >> 5;   // causal tile bound
  const __bf16* kbase = k_bf + (size_t)b * T_SEQ * D_HEAD;

  v16bf BkA[2][2], BkB[2][2];            // [sub][kc] double buffers

  // preload K operands for tile 0
#pragma unroll
  for (int sub = 0; sub < 2; ++sub) {
    const __bf16* kp = kbase + (size_t)(sub * 16 + n) * D_HEAD;
#pragma unroll
    for (int kc = 0; kc < 2; ++kc)
      BkA[sub][kc] = *(const v16bf*)(kp + kc * 32 + hi * 16);
  }

  auto tile_step = [&](int kt, v16bf (&BkCur)[2][2], v16bf (&BkNxt)[2][2]) {
    const int key0 = kt * 32;

    // ---- next tile's K operands: issued a whole iteration ahead ----------
    if (kt + 1 < nkt) {
      const int nk0 = key0 + 32;
#pragma unroll
      for (int sub = 0; sub < 2; ++sub) {
        const __bf16* kp = kbase + (size_t)(nk0 + sub * 16 + n) * D_HEAD;
#pragma unroll
        for (int kc = 0; kc < 2; ++kc)
          BkNxt[sub][kc] = *(const v16bf*)(kp + kc * 32 + hi * 16);
      }
    }

    // ---- V operands for this tile (consumed only after softmax) ----------
    v16bf Bv[DTILES];
#pragma unroll
    for (int dt = 0; dt < DTILES; ++dt) {
      const __bf16* vp = vt_bf + ((size_t)b * D_HEAD + dt * 16 + n) * T_SEQ + key0 + hi * 16;
      Bv[dt] = *(const v16bf*)vp;
    }

    // ---- prefetch tile kt+2 into WGP cache (global_prefetch_b8) ----------
    if (kt + 2 < nkt) {
      const int pk0 = key0 + 64;
      __builtin_prefetch(kbase + (size_t)(pk0 + n) * D_HEAD, 0, 3);
      __builtin_prefetch(kbase + (size_t)(pk0 + 16 + n) * D_HEAD, 0, 3);
      __builtin_prefetch(vt_bf + ((size_t)b * D_HEAD + n) * T_SEQ + pk0, 0, 3);
    }

    // ---- S = (Q/sqrt(D)) K^T : two 16x16 key sub-tiles -------------------
    v8f S[2];
#pragma unroll
    for (int sub = 0; sub < 2; ++sub) {
      v8f c;
#pragma unroll
      for (int r = 0; r < 8; ++r) c[r] = 0.f;
#pragma unroll
      for (int kc = 0; kc < 2; ++kc)
        c = __builtin_amdgcn_wmma_f32_16x16x32_bf16(false, qA[kc], false, BkCur[sub][kc],
                                                    (short)0, c, false, false);
      S[sub] = c;
    }

    // ---- causal mask (only the diagonal tile can need it) ----------------
    if (key0 + 31 > q0) {
#pragma unroll
      for (int sub = 0; sub < 2; ++sub) {
        const int key = key0 + sub * 16 + n;
#pragma unroll
        for (int r = 0; r < 8; ++r) {
          const int row = q0 + r + 8 * hi;
          if (key > row) S[sub][r] = -1.0e30f;
        }
      }
    }

    // ---- online softmax row statistics (rows span 16-lane halves) --------
    float cf[8];
#pragma unroll
    for (int r = 0; r < 8; ++r) {
      float t = fmaxf(S[0][r], S[1][r]);
      t = fmaxf(t, __shfl_xor(t, 8, 32));
      t = fmaxf(t, __shfl_xor(t, 4, 32));
      t = fmaxf(t, __shfl_xor(t, 2, 32));
      t = fmaxf(t, __shfl_xor(t, 1, 32));
      const float mn = fmaxf(m[r], t);
      cf[r] = __expf(m[r] - mn);
      m[r]  = mn;
    }

    float rs[8];
#pragma unroll
    for (int r = 0; r < 8; ++r) rs[r] = 0.f;
#pragma unroll
    for (int sub = 0; sub < 2; ++sub) {
#pragma unroll
      for (int r = 0; r < 8; ++r) {
        const float p = __expf(S[sub][r] - m[r]);
        rs[r] += p;
        Pl[w][r + 8 * hi][sub * 16 + n] = (__bf16)p;   // C/D -> LDS
      }
    }
#pragma unroll
    for (int r = 0; r < 8; ++r) {
      float t = rs[r];
      t += __shfl_xor(t, 8, 32);
      t += __shfl_xor(t, 4, 32);
      t += __shfl_xor(t, 2, 32);
      t += __shfl_xor(t, 1, 32);
      l[r] = l[r] * cf[r] + t;
#pragma unroll
      for (int dt = 0; dt < DTILES; ++dt) acc[dt][r] *= cf[r];
    }

    // ---- P back from LDS in A-layout (same-wave DS ops are in-order) -----
    v16bf PA;
    {
      const __bf16* pp = &Pl[w][n][0];
#pragma unroll
      for (int j = 0; j < 8; ++j) {
        PA[j]     = pp[hi * 16 + j];
        PA[8 + j] = pp[hi * 16 + 8 + j];
      }
    }

    // ---- acc += P @ V -----------------------------------------------------
#pragma unroll
    for (int dt = 0; dt < DTILES; ++dt)
      acc[dt] = __builtin_amdgcn_wmma_f32_16x16x32_bf16(false, PA, false, Bv[dt],
                                                        (short)0, acc[dt], false, false);
  };

  for (int kt = 0;;) {
    tile_step(kt, BkA, BkB);
    if (++kt >= nkt) break;
    tile_step(kt, BkB, BkA);
    if (++kt >= nkt) break;
  }

  // ---- z = acc / l (f32 output) ------------------------------------------
#pragma unroll
  for (int dt = 0; dt < DTILES; ++dt) {
#pragma unroll
    for (int r = 0; r < 8; ++r) {
      out[(g0 + r + 8 * hi) * D_HEAD + dt * 16 + n] = acc[dt][r] / l[r];
    }
  }
}

// ---------------------------------------------------------------------------
extern "C" void kernel_launch(void* const* d_in, const int* in_sizes, int n_in,
                              void* d_out, int out_size, void* d_ws, size_t ws_size,
                              hipStream_t stream) {
  const float* x  = (const float*)d_in[0];
  const float* Wq = (const float*)d_in[1];
  const float* bq = (const float*)d_in[2];
  const float* Wk = (const float*)d_in[3];
  const float* bk = (const float*)d_in[4];
  const float* Wv = (const float*)d_in[5];
  const float* bv = (const float*)d_in[6];
  float* out = (float*)d_out;

  // workspace: 384 KB packed weights + 3 x 2 MB bf16 Q/K/V^T  (~6.4 MB)
  char* ws = (char*)d_ws;
  __bf16* wpack = (__bf16*)ws;
  __bf16* q_bf  = (__bf16*)(ws + (384ull << 10));
  __bf16* k_bf  = (__bf16*)(ws + (384ull << 10) + (2ull << 20));
  __bf16* vt_bf = (__bf16*)(ws + (384ull << 10) + (4ull << 20));

  const int repack_elems = 3 * KSTEPS * DTILES * 512;
  repack_w_kernel<<<(repack_elems + 255) / 256, 256, 0, stream>>>(Wq, Wk, Wv, wpack);
  proj_kernel<<<768, 128, 0, stream>>>(x, wpack, bq, bk, bv, q_bf, k_bf, vt_bf);
  attn_kernel<<<256, 128, 0, stream>>>(q_bf, k_bf, vt_bf, out);
}
